// TransitionProbLayer_70798240907657
// MI455X (gfx1250) — matrix operand order repaired and checked by reading, not compiled, not run on hardware
//
#include <hip/hip_runtime.h>

// ---------------------------------------------------------------------------
// TransitionProbLayer for MI455X (gfx1250, wave32, WMMA).
//
// GEMM: [B*T=46592, K=64] x [64, N=961] via v_wmma_f32_16x16x32_bf16,
// fused with exp / row-normalize (shfl_xor tree over 16 lanes) /
// diagonal-insert scatter / absorbing row / max(.,BASIS).
//
// N is padded per i-group: each i in [0,31) owns 32 columns (j=0..30 real,
// j=31 zero pad) = two 16-wide WMMA N-tiles. B fragments (bf16, pre-swizzled
// to the wave32 WMMA B layout) and padded -mu live in LDS; the inner loop is
// ds_load_b128 + 4x WMMA + exp/reduce/scatter with non-temporal stores.
// ---------------------------------------------------------------------------

typedef __attribute__((ext_vector_type(16))) __bf16 v16bf;
typedef __attribute__((ext_vector_type(8)))  float  v8f;

#define N_STATES 32
#define SM1      31
#define N_CAMP   64
#define DAYS     365
#define T_DAYS   364
#define BATCH    128
#define BASISF   1e-10f
#define NTILES   62                    // 31 i-groups * 2 half-tiles (16 cols each)
#define NFRAGS   (NTILES * 2)          // * 2 k-blocks (K=0..31, 32..63) = 124
#define B_ELEMS  (NFRAGS * 32 * 16)    // bf16 elements in the fragment pool
#define WS_BYTES ((size_t)B_ELEMS * 2) // 126976 bytes

// B-fragment element mapping used here (contiguous k, not AMD's table order) (B is 32x16, KxN):
//   lane l, element e:  K = e + 16*(l>>4),  N(column) = l & 15
// A-fragment element mapping (A is 16x32, MxK):
//   lane l, element e:  M = l & 15,  K = e + 16*(l>>4) (same k order as B)

__global__ __launch_bounds__(256)
void prep_beta_kernel(const float* __restrict__ beta, __bf16* __restrict__ wsB) {
  int tid = blockIdx.x * blockDim.x + threadIdx.x;     // one thread = one (frag,lane)
  if (tid >= NFRAGS * 32) return;
  int f     = tid >> 5;
  int lane  = tid & 31;
  int ntile = f >> 1;                  // 0..61
  int kb    = f & 1;                   // k-block: 0 -> K 0..31, 1 -> K 32..63
  int i     = ntile >> 1;              // i-group 0..30
  int h     = ntile & 1;               // half: columns 0..15 or 16..31 of i-group
  int j     = 16 * h + (lane & 15);    // padded column index within i-group
  int hf    = lane >> 4;
  // branch-free: clamp index, multiply by mask
  float msk = (j < SM1) ? 1.0f : 0.0f;
  int   jm  = (j < SM1) ? j : 0;
  __bf16* dst = wsB + (size_t)tid * 16;
#pragma unroll
  for (int e = 0; e < 16; ++e) {
    int c = e + 16 * hf + 32 * kb;     // contraction index (campaign)
    float v = beta[(size_t)c * (SM1 * SM1) + i * SM1 + jm] * msk;
    dst[e] = (__bf16)v;
  }
}

__global__ __launch_bounds__(256)
void transition_kernel(const float* __restrict__ adstock,
                       const float* __restrict__ mu,
                       const float* __restrict__ beta,
                       const __bf16* __restrict__ wsB,
                       float* __restrict__ out,
                       int use_ws) {
  __shared__ alignas(32) __bf16 Blds[B_ELEMS];     // 126976 B: pre-swizzled B frags
  __shared__ float MuLds[NTILES * 16];             //   3968 B: padded -mu per tile col

  const int lane = threadIdx.x & 31;
  const int warp = threadIdx.x >> 5;
  const int hf   = lane >> 4;
  const int lc   = lane & 15;

  // ---- One-time LDS fill (whole workgroup) ----
  if (use_ws) {
    const uint4* src = (const uint4*)wsB;
    uint4* dst = (uint4*)Blds;
#pragma unroll 4
    for (int k = threadIdx.x; k < (int)(WS_BYTES / 16); k += 256) dst[k] = src[k];
  } else {
    for (int lf = threadIdx.x; lf < NFRAGS * 32; lf += 256) {
      int f     = lf >> 5;
      int ln    = lf & 31;
      int ntile = f >> 1;
      int kb    = f & 1;
      int i     = ntile >> 1;
      int h     = ntile & 1;
      int j     = 16 * h + (ln & 15);
      int lhf   = ln >> 4;
      float msk = (j < SM1) ? 1.0f : 0.0f;
      int   jm  = (j < SM1) ? j : 0;
      __bf16* d = Blds + (size_t)lf * 16;
#pragma unroll
      for (int e = 0; e < 16; ++e) {
        int c = e + 16 * lhf + 32 * kb;
        d[e] = (__bf16)(beta[(size_t)c * (SM1 * SM1) + i * SM1 + jm] * msk);
      }
    }
  }
  for (int idx = threadIdx.x; idx < NTILES * 16; idx += 256) {
    int ntile = idx >> 4;
    int colc  = idx & 15;
    int i     = ntile >> 1;
    int j     = 16 * (ntile & 1) + colc;
    int jm    = (j < SM1) ? j : 0;
    float v   = -mu[i * SM1 + jm];
    MuLds[idx] = (j < SM1) ? v : 0.0f;
  }
  __syncthreads();

  // ---- Load both A fragments (this wave's 16 rows x K=64, bf16) once ----
  const int mtile  = blockIdx.x * 8 + warp;  // 2912 M-tiles of 16 (b,t) rows
  const int mrow_a = mtile * 16 + lc;        // A layout: M = lane&15
  const int bb = mrow_a / T_DAYS;
  const int tt = mrow_a % T_DAYS;
  const float* aptr = adstock + (size_t)bb * (N_CAMP * DAYS) + (tt + 1);
  v16bf a0, a1;
#pragma unroll
  for (int e = 0; e < 16; ++e) {
    int K = e + 16 * hf;
    a0[e] = (__bf16)aptr[(size_t)K * DAYS];          // c = K
    a1[e] = (__bf16)aptr[(size_t)(K + 32) * DAYS];   // c = K + 32
  }

  const int jc0 = lc;                 // padded-column index in tile 0 (0..15)
  const int jc1 = 16 + lc;            // padded-column index in tile 1 (16..31)
  const float m1    = (lc < 15) ? 1.0f : 0.0f;   // jc1 < 31  <=>  lc < 15
  const bool  valid1 = (lc < 15);
  const size_t row_base = (size_t)mtile * 16 + 8 * hf;  // C rows: r + 8*(lane>>4)
  const v16bf* Bf = (const v16bf*)Blds;

  for (int i = 0; i < SM1; ++i) {
    const int nt0 = 2 * i, nt1 = 2 * i + 1;
    v16bf b00 = Bf[(nt0 * 2 + 0) * 32 + lane];
    v16bf b01 = Bf[(nt0 * 2 + 1) * 32 + lane];
    v16bf b10 = Bf[(nt1 * 2 + 0) * 32 + lane];
    v16bf b11 = Bf[(nt1 * 2 + 1) * 32 + lane];

    v8f acc0 = {}, acc1 = {};
    acc0 = __builtin_amdgcn_wmma_f32_16x16x32_bf16(false, a0, false, b00, (short)0, acc0, false, false);
    acc0 = __builtin_amdgcn_wmma_f32_16x16x32_bf16(false, a1, false, b01, (short)0, acc0, false, false);
    acc1 = __builtin_amdgcn_wmma_f32_16x16x32_bf16(false, a0, false, b10, (short)0, acc1, false, false);
    acc1 = __builtin_amdgcn_wmma_f32_16x16x32_bf16(false, a1, false, b11, (short)0, acc1, false, false);

    const float mlo = MuLds[nt0 * 16 + lc];
    const float mhi = MuLds[nt1 * 16 + lc];
    const int col0 = (jc0 < i) ? jc0 : jc0 + 1;                 // diagonal-insert shift
    const int col1 = valid1 ? ((jc1 < i) ? jc1 : jc1 + 1) : i;  // pad lane -> column i

#pragma unroll
    for (int r = 0; r < 8; ++r) {
      float n0 = __expf(acc0[r] + mlo);
      float n1 = __expf(acc1[r] + mhi);
      float s  = n0 + n1 * m1;
      // reduce over the 16 columns of each half-wave (rows are per-half)
      s += __shfl_xor(s, 1);
      s += __shfl_xor(s, 2);
      s += __shfl_xor(s, 4);
      s += __shfl_xor(s, 8);
      float rden = __builtin_amdgcn_rcpf(1.0f + s);   // den = 1 + sum(num); Q_same = 1/den
      float q0 = fmaxf(n0 * rden, BASISF);
      float q1 = fmaxf((valid1 ? n1 : 1.0f) * rden, BASISF);
      size_t base = (row_base + r) * (N_STATES * N_STATES) + (size_t)i * N_STATES;
      __builtin_nontemporal_store(q0, &out[base + col0]);
      __builtin_nontemporal_store(q1, &out[base + col1]);
    }
  }

  // ---- Absorbing last row (i = 31) ----
  const float lastv = 1.0f - (N_STATES - 1) * BASISF;
#pragma unroll
  for (int r = 0; r < 8; ++r) {
    size_t base = (row_base + r) * (N_STATES * N_STATES) + (size_t)SM1 * N_STATES;
    __builtin_nontemporal_store(BASISF, &out[base + jc0]);
    __builtin_nontemporal_store((jc1 == 31) ? lastv : BASISF, &out[base + jc1]);
  }
}

extern "C" void kernel_launch(void* const* d_in, const int* in_sizes, int n_in,
                              void* d_out, int out_size, void* d_ws, size_t ws_size,
                              hipStream_t stream) {
  const float* adstock = (const float*)d_in[0];   // [128, 64, 365] f32
  const float* mu      = (const float*)d_in[1];   // [961] f32
  const float* beta    = (const float*)d_in[2];   // [64*961] f32
  float* out = (float*)d_out;                     // [128, 364, 32, 32] f32
  __bf16* wsB = (__bf16*)d_ws;

  const int use_ws = (d_ws != nullptr && ws_size >= WS_BYTES) ? 1 : 0;
  if (use_ws) {
    prep_beta_kernel<<<(NFRAGS * 32 + 255) / 256, 256, 0, stream>>>(beta, wsB);
  }
  // 2912 M-tiles, 8 waves (tiles) per 256-thread block -> 364 blocks
  transition_kernel<<<(BATCH * T_DAYS / 16) / 8, 256, 0, stream>>>(
      adstock, mu, beta, wsB, out, use_ws);
}
